// GAT_GCN_30485677867034
// MI455X (gfx1250) — compile-verified, matrix operand order checked by reading, not run on hardware
//
#include <hip/hip_runtime.h>
#include <math.h>

// Problem constants (match reference)
#define N_NODES 20000
#define N_EDGES 160000
#define NG      128
#define NF      114
#define NH      10
#define NHF     1140            // NF*NH
#define EPAD    (N_EDGES + N_NODES)   // edges + self loops
#define F2      (2*NHF)         // 2280

typedef __attribute__((ext_vector_type(16))) _Float16 v16h;
typedef __attribute__((ext_vector_type(8)))  _Float16 v8h;
typedef __attribute__((ext_vector_type(8)))  float    v8f;

// ---------------------------------------------------------------------------
// Helpers
// ---------------------------------------------------------------------------
__device__ inline void atomicMaxF(float* addr, float val) {
  int old = __float_as_int(*addr);
  while (__int_as_float(old) < val) {
    int prev = atomicCAS((int*)addr, old, __float_as_int(val));
    if (prev == old) break;
    old = prev;
  }
}

__device__ inline void edge_sd(const int* __restrict__ ei, int eid, int& s, int& d) {
  if (eid < N_EDGES) { s = ei[eid]; d = ei[N_EDGES + eid]; }
  else               { s = d = eid - N_EDGES; }       // self loops appended
}

__global__ void fill_f32(float* __restrict__ p, int n, float v) {
  int i = blockIdx.x * blockDim.x + threadIdx.x;
  if (i < n) p[i] = v;
}

// ---------------------------------------------------------------------------
// f32 -> f16 staging (hoists conversion + padding out of the WMMA loop)
// ---------------------------------------------------------------------------
// A[M,K] (row-major, leading dim lda) -> A16[M,Kpad] f16, zero K-padding
__global__ void cvt_a_f16(const float* __restrict__ A, int lda, int M, int K,
                          _Float16* __restrict__ out, int Kpad) {
  long i = (long)blockIdx.x * blockDim.x + threadIdx.x;
  if (i >= (long)M * Kpad) return;
  int r = (int)(i / Kpad), c = (int)(i % Kpad);
  out[i] = (c < K) ? (_Float16)A[(size_t)r * lda + c] : (_Float16)0.f;
}
// B[K,N] (row-major) -> Bt16[Npad,Kpad] f16 transposed, zero-padded both dims
// (Npad is a multiple of 64 so the 4-tile register-blocked GEMM never reads OOB)
__global__ void cvt_bt_f16(const float* __restrict__ B, int K, int N,
                           _Float16* __restrict__ out, int Kpad, int Npad) {
  long i = (long)blockIdx.x * blockDim.x + threadIdx.x;
  if (i >= (long)Npad * Kpad) return;
  int n = (int)(i / Kpad), k = (int)(i % Kpad);
  out[i] = (n < N && k < K) ? (_Float16)B[(size_t)k * N + n] : (_Float16)0.f;
}

__device__ inline v16h frag16(const _Float16* p) {
  v8h lo = *(const v8h*)(p);
  v8h hi = *(const v8h*)(p + 16);
  return __builtin_shufflevector(lo, hi, 0,1,2,3,4,5,6,7,8,9,10,11,12,13,14,15);
}

// ---------------------------------------------------------------------------
// Register-blocked WMMA GEMM on pre-staged f16:
//   C[M,N] = act(A16[M,Kpad] * Bt16[Npad,Kpad]^T + bias)
// Each wave owns a 16x64 C strip: 4 accumulators share one A fragment per
// k-step (10 b128 loads : 4 WMMAs). 4 waves/block cover 256 columns.
// Wave32, EXEC all-ones, branch-free before the WMMAs.
// Fragment layouts per CDNA5 ISA 7.12.2:
//   A 16x32 f16 : lanes 0-15 row M=l, halves[0..7]=K0..7, [8..15]=K16..23;
//                 lanes 16-31 same rows, K+8 / K+24.  (contiguous 8-half runs)
//   B 32x16 f16 : lane -> column (l&15), same K striping (contiguous in Bt16).
//   C 16x16 f32 : acc[i] -> M = i + 8*(lane>=16), N = lane&15.
// M is a multiple of 16, Kpad of 32, Npad of 64; only C stores guard n < N.
// ---------------------------------------------------------------------------
__global__ void wmma_gemm16(const _Float16* __restrict__ A16,
                            const _Float16* __restrict__ Bt16,
                            float* __restrict__ C, int ldc,
                            int Nc, int Npad, int Kpad,
                            const float* __restrict__ bias, int act) {
  int wave = threadIdx.x >> 5;
  int lane = threadIdx.x & 31;
  int n0 = (blockIdx.x * 4 + wave) * 64;      // 64-wide strip per wave
  int m0 = blockIdx.y * 16;
  if (n0 >= Npad) return;                     // whole wave exits together

  int hi   = lane >> 4;                       // lane half (0/1)
  int l15  = lane & 15;
  const _Float16* ap = A16  + (size_t)(m0 + l15) * Kpad + hi * 8;
  const _Float16* bp = Bt16 + (size_t)(n0 + l15) * Kpad + hi * 8;
  const size_t bstep = (size_t)16 * Kpad;     // next 16-column tile

  v8f acc0 = {}, acc1 = {}, acc2 = {}, acc3 = {};
  for (int k0 = 0; k0 < Kpad; k0 += 32) {
    v16h af = frag16(ap + k0);
    v16h b0 = frag16(bp + k0);
    v16h b1 = frag16(bp + k0 + bstep);
    v16h b2 = frag16(bp + k0 + 2 * bstep);
    v16h b3 = frag16(bp + k0 + 3 * bstep);
    acc0 = __builtin_amdgcn_wmma_f32_16x16x32_f16(false, af, false, b0, (short)0, acc0, false, false);
    acc1 = __builtin_amdgcn_wmma_f32_16x16x32_f16(false, af, false, b1, (short)0, acc1, false, false);
    acc2 = __builtin_amdgcn_wmma_f32_16x16x32_f16(false, af, false, b2, (short)0, acc2, false, false);
    acc3 = __builtin_amdgcn_wmma_f32_16x16x32_f16(false, af, false, b3, (short)0, acc3, false, false);
  }

  v8f accs[4] = {acc0, acc1, acc2, acc3};
#pragma unroll
  for (int j = 0; j < 4; ++j) {
#pragma unroll
    for (int i = 0; i < 8; ++i) {
      int m = m0 + hi * 8 + i;
      int n = n0 + j * 16 + l15;
      if (n < Nc) {
        float v = accs[j][i];
        if (bias) v += bias[n];
        if (act)  v = v > 0.f ? v : 0.01f * v;   // LeakyReLU(0.01)
        C[(size_t)m * ldc + n] = v;
      }
    }
  }
}

// ---------------------------------------------------------------------------
// GAT attention pieces
// ---------------------------------------------------------------------------
__global__ void attn_scores(const float* __restrict__ h1,
                            const float* __restrict__ att_src,
                            const float* __restrict__ att_dst,
                            float* __restrict__ aS, float* __restrict__ aD) {
  int idx = blockIdx.x * blockDim.x + threadIdx.x;   // n*NH + h
  if (idx >= N_NODES * NH) return;
  int n = idx / NH, h = idx % NH;
  const float* hp = h1 + (size_t)n * NHF + h * NF;
  const float* as = att_src + h * NF;
  const float* ad = att_dst + h * NF;
  float s = 0.f, d = 0.f;
  for (int f = 0; f < NF; ++f) { float v = hp[f]; s += v * as[f]; d += v * ad[f]; }
  aS[idx] = s; aD[idx] = d;
}

__global__ void edge_scores(const int* __restrict__ ei,
                            const float* __restrict__ aS, const float* __restrict__ aD,
                            float* __restrict__ e, float* __restrict__ m) {
  int i = blockIdx.x * blockDim.x + threadIdx.x;     // eid*NH + h
  if (i >= EPAD * NH) return;
  int eid = i / NH, h = i % NH;
  int s, d; edge_sd(ei, eid, s, d);
  float v = aS[s * NH + h] + aD[d * NH + h];
  v = v > 0.f ? v : 0.2f * v;                        // LeakyReLU(0.2)
  e[i] = v;
  atomicMaxF(&m[d * NH + h], v);
}

__global__ void edge_expden(const int* __restrict__ ei,
                            float* __restrict__ e, const float* __restrict__ m,
                            float* __restrict__ den) {
  int i = blockIdx.x * blockDim.x + threadIdx.x;
  if (i >= EPAD * NH) return;
  int eid = i / NH, h = i % NH;
  int s, d; edge_sd(ei, eid, s, d);
  float ex = expf(e[i] - m[d * NH + h]);
  e[i] = ex;
  atomicAdd(&den[d * NH + h], ex);
}

// out[dst, :] += alpha_h * h1[src, :]    (block per edge, coalesced rows)
__global__ void gat_agg(const int* __restrict__ ei,
                        const float* __restrict__ ex, const float* __restrict__ den,
                        const float* __restrict__ h1, float* __restrict__ out) {
  __shared__ float alpha[NH];
  int eid = blockIdx.x;
  int s, d; edge_sd(ei, eid, s, d);
  if (threadIdx.x < NH)
    alpha[threadIdx.x] = ex[(size_t)eid * NH + threadIdx.x] /
                         den[(size_t)d * NH + threadIdx.x];
  __syncthreads();
  const float* hs = h1 + (size_t)s * NHF;
  float* od = out + (size_t)d * NHF;
  for (int f = threadIdx.x; f < NHF; f += blockDim.x)
    atomicAdd(&od[f], alpha[f / NF] * hs[f]);
}

__global__ void bias_act_rows(float* __restrict__ x, const float* __restrict__ b,
                              int total, int cols) {
  int i = blockIdx.x * blockDim.x + threadIdx.x;
  if (i >= total) return;
  float v = x[i] + b[i % cols];
  x[i] = v > 0.f ? v : 0.01f * v;
}

// ---------------------------------------------------------------------------
// GCN pieces
// ---------------------------------------------------------------------------
__global__ void deg_count(const int* __restrict__ ei, float* __restrict__ deg) {
  int eid = blockIdx.x * blockDim.x + threadIdx.x;
  if (eid >= EPAD) return;
  int s, d; edge_sd(ei, eid, s, d);
  atomicAdd(&deg[d], 1.0f);
}
__global__ void deg_rsqrt(float* __restrict__ deg) {
  int n = blockIdx.x * blockDim.x + threadIdx.x;
  if (n < N_NODES) deg[n] = rsqrtf(fmaxf(deg[n], 1.0f));
}
__global__ void gcn_agg(const int* __restrict__ ei, const float* __restrict__ dinv,
                        const float* __restrict__ hlin, float* __restrict__ out) {
  int eid = blockIdx.x;
  int s, d; edge_sd(ei, eid, s, d);
  float c = dinv[s] * dinv[d];
  const float* hs = hlin + (size_t)s * NHF;
  float* od = out + (size_t)d * NHF;
  for (int f = threadIdx.x; f < NHF; f += blockDim.x)
    atomicAdd(&od[f], c * hs[f]);
}

// ---------------------------------------------------------------------------
// Pooling
// ---------------------------------------------------------------------------
__global__ void pool_kernel(const float* __restrict__ h, const int* __restrict__ batch,
                            float* __restrict__ mx, float* __restrict__ sm,
                            float* __restrict__ cnt) {
  int n = blockIdx.x;
  int g = batch[n];
  if (threadIdx.x == 0) atomicAdd(&cnt[g], 1.0f);
  const float* hp = h + (size_t)n * NHF;
  float* mg = mx + (size_t)g * NHF;
  float* sg = sm + (size_t)g * NHF;
  for (int f = threadIdx.x; f < NHF; f += blockDim.x) {
    float v = hp[f];
    atomicMaxF(&mg[f], v);
    atomicAdd(&sg[f], v);
  }
}
__global__ void pool_fin(const float* __restrict__ mx, const float* __restrict__ sm,
                         const float* __restrict__ cnt, float* __restrict__ g) {
  int i = blockIdx.x * blockDim.x + threadIdx.x;
  if (i >= NG * NHF) return;
  int gi = i / NHF, c = i % NHF;
  float mv = mx[i];
  if (!isfinite(mv)) mv = 0.f;
  float cn = fmaxf(cnt[gi], 1.0f);
  g[(size_t)gi * F2 + c]       = mv;
  g[(size_t)gi * F2 + NHF + c] = sm[i] / cn;
}

// ---------------------------------------------------------------------------
// Host side
// ---------------------------------------------------------------------------
struct Scratch {
  float *bufA, *bufB, *aS, *aD, *eB, *mB, *dB, *degB;
  float *mxB, *smB, *cntB, *gB, *g1B, *xcB, *fc1B, *fc2B;
  _Float16 *A16, *B16;
};

// C[M,N] = act(A[M,K] @ B[K,N] + bias); M must be a multiple of 16.
static inline void gemm(const float* A, int lda, const float* B,
                        float* C, int ldc, int M, int N, int K,
                        const float* bias, int act, const Scratch& S, hipStream_t s) {
  int Kpad = (K + 31) & ~31;
  int Npad = (N + 63) & ~63;                  // multiple of 64 for 4-tile strips
  long at = (long)M * Kpad;
  cvt_a_f16<<<dim3((unsigned)((at + 255) / 256)), 256, 0, s>>>(A, lda, M, K, S.A16, Kpad);
  long bt = (long)Npad * Kpad;
  cvt_bt_f16<<<dim3((unsigned)((bt + 255) / 256)), 256, 0, s>>>(B, K, N, S.B16, Kpad, Npad);
  dim3 grid((Npad + 255) / 256, M / 16);      // 4 waves x 64 cols per block
  wmma_gemm16<<<grid, dim3(128), 0, s>>>(S.A16, S.B16, C, ldc, N, Npad, Kpad, bias, act);
}

static inline void fill(float* p, int n, float v, hipStream_t s) {
  fill_f32<<<(n + 255) / 256, 256, 0, s>>>(p, n, v);
}

static void run_branch(const float* x, const int* ei, const int* batch,
                       const float* const* P,  // gatW,attS,attD,gatb,gcnW,gcnb,f1W,f1b,f2W,f2b
                       const Scratch& S, float* xcCol, hipStream_t st) {
  const int NHtot = N_NODES * NH, EHtot = EPAD * NH, NHFtot = N_NODES * NHF;
  // 1) h1 = x @ gatW -> bufA   (bias applied after aggregation, per reference)
  gemm(x, NF, P[0], S.bufA, NHF, N_NODES, NHF, NF, nullptr, 0, S, st);
  // 2) per-node attention scores
  attn_scores<<<(NHtot + 255) / 256, 256, 0, st>>>(S.bufA, P[1], P[2], S.aS, S.aD);
  // 3) edge softmax (max, exp, den)
  fill(S.mB, NHtot, -INFINITY, st);
  fill(S.dB, NHtot, 0.f, st);
  edge_scores<<<(EHtot + 255) / 256, 256, 0, st>>>(ei, S.aS, S.aD, S.eB, S.mB);
  edge_expden<<<(EHtot + 255) / 256, 256, 0, st>>>(ei, S.eB, S.mB, S.dB);
  // 4) GAT aggregation -> bufB; then +bias, LeakyReLU(0.01)
  fill(S.bufB, NHFtot, 0.f, st);
  gat_agg<<<EPAD, 256, 0, st>>>(ei, S.eB, S.dB, S.bufA, S.bufB);
  bias_act_rows<<<(NHFtot + 255) / 256, 256, 0, st>>>(S.bufB, P[3], NHFtot, NHF);
  // 5) GCN linear: bufA = bufB @ gcnW
  gemm(S.bufB, NHF, P[4], S.bufA, NHF, N_NODES, NHF, NHF, nullptr, 0, S, st);
  // 6) degree norm + GCN aggregation -> bufB; +bias, LeakyReLU
  fill(S.degB, N_NODES, 0.f, st);
  deg_count<<<(EPAD + 255) / 256, 256, 0, st>>>(ei, S.degB);
  deg_rsqrt<<<(N_NODES + 255) / 256, 256, 0, st>>>(S.degB);
  fill(S.bufB, NHFtot, 0.f, st);
  gcn_agg<<<EPAD, 256, 0, st>>>(ei, S.degB, S.bufA, S.bufB);
  bias_act_rows<<<(NHFtot + 255) / 256, 256, 0, st>>>(S.bufB, P[5], NHFtot, NHF);
  // 7) pooling: gB[G, 2*HF] = [max || mean]
  fill(S.mxB, NG * NHF, -INFINITY, st);
  fill(S.smB, NG * NHF, 0.f, st);
  fill(S.cntB, NG, 0.f, st);
  pool_kernel<<<N_NODES, 256, 0, st>>>(S.bufB, batch, S.mxB, S.smB, S.cntB);
  pool_fin<<<(NG * NHF + 255) / 256, 256, 0, st>>>(S.mxB, S.smB, S.cntB, S.gB);
  // 8) g1 = lrelu(g @ f1W + f1b); branch out = g1 @ f2W + f2b -> xc columns
  gemm(S.gB, F2, P[6], S.g1B, 1000, NG, 1000, F2, P[7], 1, S, st);
  gemm(S.g1B, 1000, P[8], xcCol, 256, NG, 64, 1000, P[9], 0, S, st);
}

extern "C" void kernel_launch(void* const* d_in, const int* in_sizes, int n_in,
                              void* d_out, int out_size, void* d_ws, size_t ws_size,
                              hipStream_t stream) {
  (void)in_sizes; (void)n_in; (void)out_size; (void)ws_size;
  const float* x1     = (const float*)d_in[0];
  const int*   ei1    = (const int*)  d_in[1];
  const int*   b1     = (const int*)  d_in[2];
  const float* x2     = (const float*)d_in[3];
  const int*   ei2    = (const int*)  d_in[4];
  const int*   b2     = (const int*)  d_in[5];
  const float* target = (const float*)d_in[6];
  const float* P1[10]; for (int i = 0; i < 10; ++i) P1[i] = (const float*)d_in[7 + i];
  const float* P2[10]; for (int i = 0; i < 10; ++i) P2[i] = (const float*)d_in[17 + i];
  const float* fcxtW = (const float*)d_in[27]; const float* fcxtb = (const float*)d_in[28];
  const float* fc1W  = (const float*)d_in[29]; const float* fc1b  = (const float*)d_in[30];
  const float* fc2W  = (const float*)d_in[31]; const float* fc2b  = (const float*)d_in[32];
  const float* outW  = (const float*)d_in[33]; const float* outb  = (const float*)d_in[34];

  char* base = (char*)d_ws;
  size_t off = 0;
  auto carveB = [&](size_t bytes) {
    char* p = base + off;
    off += (bytes + 255) & ~(size_t)255;
    return (void*)p;
  };
  Scratch S;
  S.bufA = (float*)carveB(sizeof(float) * (size_t)N_NODES * NHF);
  S.bufB = (float*)carveB(sizeof(float) * (size_t)N_NODES * NHF);
  S.aS   = (float*)carveB(sizeof(float) * (size_t)N_NODES * NH);
  S.aD   = (float*)carveB(sizeof(float) * (size_t)N_NODES * NH);
  S.eB   = (float*)carveB(sizeof(float) * (size_t)EPAD * NH);
  S.mB   = (float*)carveB(sizeof(float) * (size_t)N_NODES * NH);
  S.dB   = (float*)carveB(sizeof(float) * (size_t)N_NODES * NH);
  S.degB = (float*)carveB(sizeof(float) * (size_t)N_NODES);
  S.mxB  = (float*)carveB(sizeof(float) * (size_t)NG * NHF);
  S.smB  = (float*)carveB(sizeof(float) * (size_t)NG * NHF);
  S.cntB = (float*)carveB(sizeof(float) * (size_t)NG);
  S.gB   = (float*)carveB(sizeof(float) * (size_t)NG * F2);
  S.g1B  = (float*)carveB(sizeof(float) * (size_t)NG * 1000);
  S.xcB  = (float*)carveB(sizeof(float) * (size_t)NG * 256);
  S.fc1B = (float*)carveB(sizeof(float) * (size_t)NG * 128);
  S.fc2B = (float*)carveB(sizeof(float) * (size_t)NG * 32);
  // f16 staging: A16 max of (20000 x 1152) and (128 x 2304); B16 max 1024 x 2304
  S.A16  = (_Float16*)carveB(sizeof(_Float16) * (size_t)N_NODES * 1152);
  S.B16  = (_Float16*)carveB(sizeof(_Float16) * (size_t)1024 * 2304);

  // Two GNN branches -> xc[:, 0:64] and xc[:, 64:128]
  run_branch(x1, ei1, b1, P1, S, S.xcB + 0,  stream);
  run_branch(x2, ei2, b2, P2, S, S.xcB + 64, stream);
  // Target head -> xc[:, 128:256]
  gemm(target, 1000, fcxtW, S.xcB + 128, 256, NG, 128, 1000, fcxtb, 0, S, stream);
  // Final MLP head
  gemm(S.xcB, 256, fc1W, S.fc1B, 128, NG, 128, 256, fc1b, 1, S, stream);
  gemm(S.fc1B, 128, fc2W, S.fc2B, 32, NG, 32, 128, fc2b, 1, S, stream);
  gemm(S.fc2B, 32, outW, (float*)d_out, 1, NG, 1, 32, outb, 0, S, stream);
}
